// GraphNetwork_214748364989
// MI455X (gfx1250) — matrix operand order repaired and checked by reading, not compiled, not run on hardware
//
#include <hip/hip_runtime.h>

// ---------------- problem constants ----------------
#define B_    2
#define N_    32768
#define E_    32768
#define H_    32          // hidden width
#define NL_   32          // number of scanned hidden layers per MLP
#define TPB   256         // 8 wave32 per block; each wave owns 16 elements
#define MS    36864       // packed-weight stride per MLP (halves): Win 2048 | Wh 32768 | Wout 512

typedef _Float16 v16h __attribute__((ext_vector_type(16)));
typedef _Float16 v8h  __attribute__((ext_vector_type(8)));
typedef _Float16 v2h  __attribute__((ext_vector_type(2)));
typedef float    v8f  __attribute__((ext_vector_type(8)));
typedef float    v4f  __attribute__((ext_vector_type(4)));
typedef int      v8i  __attribute__((ext_vector_type(8)));

// D = A(16x32 f16) x B(32x16 f16) + C(16x16 f32)
// We compute Y^T = W^T * X^T : A = packed W^T fragment, B = activations X^T.
__device__ __forceinline__ v8f wmma16x16x32(v16h a, v16h b, v8f c) {
  return __builtin_amdgcn_wmma_f32_16x16x32_f16(false, a, false, b, (short)0, c,
                                                false, false);
}

// lane <-> lane^16 exchange as pure VALU (v_permlanex16_b32, identity sel).
__device__ __forceinline__ int xl16(int v) {
  return __builtin_amdgcn_permlanex16(v, v, 0x76543210, 0xfedcba98, false,
                                      false);
}
__device__ __forceinline__ float xl16f(float v) {
  return __builtin_bit_cast(float, xl16(__builtin_bit_cast(int, v)));
}

// pack two f32 -> one dword of two f16 (v_cvt_pk_f16_f32), register-resident
__device__ __forceinline__ int pkh(float a, float b) {
  v2h p;
  p[0] = (_Float16)a;
  p[1] = (_Float16)b;
  return __builtin_bit_cast(int, p);
}

// A fragment (W^T) from packed f16 buffer: per-lane 32 contiguous bytes.
__device__ __forceinline__ v16h load_frag(const _Float16* __restrict__ frag,
                                          int fi, int lane) {
  const _Float16* p = frag + ((size_t)(fi * 32 + lane) << 4);
  const v8h lo = *(const v8h*)p;                    // global_load_b128
  const v8h hi = *(const v8h*)(p + 8);              // global_load_b128
  return __builtin_shufflevector(lo, hi, 0, 1, 2, 3, 4, 5, 6, 7, 8, 9, 10, 11,
                                 12, 13, 14, 15);
}

// B fragment (activations X^T) from a 16-float global row: lane holds 16
// consecutive features of element n0.
__device__ __forceinline__ v16h load16f(const float* __restrict__ src) {
  const v4f* s = (const v4f*)src;                   // 4x global_load_b128
  const v4f a = s[0], b = s[1], c = s[2], d = s[3];
  v16h x;
#pragma unroll
  for (int i = 0; i < 4; ++i) {
    x[i] = (_Float16)a[i];
    x[4 + i] = (_Float16)b[i];
    x[8 + i] = (_Float16)c[i];
    x[12 + i] = (_Float16)d[i];
  }
  return x;
}

// Pack fp32 weights (stacked layers along K) into f16 A-fragment order for
// A = W^T. Frag fi = kt*2 + t: lane m-index = t*16 + (lane&15) (= out feature),
// K = kt*32 + j + 16*(lane>=16) (= in feature). A[M][K] = W[K][M].
__global__ __launch_bounds__(TPB) void k_pack(const float* __restrict__ W,
                                              _Float16* __restrict__ dst,
                                              int ld, int ktotal, int nmax,
                                              int nfrag) {
  const int tid = blockIdx.x * TPB + threadIdx.x;
  const int fi = tid >> 5, lane = tid & 31;
  if (fi >= nfrag) return;
  const int kt = fi >> 1, t = fi & 1;
  const int m = (t << 4) + (lane & 15);
  const int kh = (lane < 16) ? 0 : 16;
  v8h h0, h1;
#pragma unroll
  for (int j = 0; j < 8; ++j) {
    const int ka = kt * 32 + j + kh;        // halves j=0..7  -> K = j + kh
    const int kb = kt * 32 + 8 + j + kh;    // halves j=8..15 -> K = 8 + j + kh
    h0[j] = (_Float16)((ka < ktotal && m < nmax) ? W[(size_t)ka * ld + m] : 0.0f);
    h1[j] = (_Float16)((kb < ktotal && m < nmax) ? W[(size_t)kb * ld + m] : 0.0f);
  }
  _Float16* p = dst + ((size_t)(fi * 32 + lane) << 4);
  *(v8h*)p = h0;
  *(v8h*)(p + 8) = h1;
}

// Epilogue: bias + relu (+ LayerNorm via permlanex16) then repack the two C
// tiles into the next layer's B fragment with a single lane<->lane^16 swap.
// C layout: lane holds elem col n0=lane&15; ct = feats mofs..mofs+7,
// cb = feats 16+mofs..16+mofs+7, mofs = 8*(lane>=16).
template <bool LN>
__device__ __forceinline__ v16h epilogue(v8f ct, v8f cb,
                                         const float* __restrict__ bias,
                                         const float* __restrict__ g,
                                         const float* __restrict__ bb,
                                         int lane) {
  const int mofs = (lane < 16) ? 0 : 8;
  const v4f bt0 = *(const v4f*)(bias + mofs);
  const v4f bt1 = *(const v4f*)(bias + mofs + 4);
  const v4f bo0 = *(const v4f*)(bias + 16 + mofs);
  const v4f bo1 = *(const v4f*)(bias + 16 + mofs + 4);
  float vt[8], vb[8];
#pragma unroll
  for (int r = 0; r < 4; ++r) {
    vt[r] = fmaxf(ct[r] + bt0[r], 0.0f);
    vt[4 + r] = fmaxf(ct[4 + r] + bt1[r], 0.0f);
    vb[r] = fmaxf(cb[r] + bo0[r], 0.0f);
    vb[4 + r] = fmaxf(cb[4 + r] + bo1[r], 0.0f);
  }
  if (LN) {  // element's 32 features live in lanes {l, l^16}
    float s = 0.0f, sq = 0.0f;
#pragma unroll
    for (int r = 0; r < 8; ++r) {
      s += vt[r] + vb[r];
      sq += vt[r] * vt[r] + vb[r] * vb[r];
    }
    s += xl16f(s);
    sq += xl16f(sq);
    const float mu = s * (1.0f / 32.0f);
    const float rs = rsqrtf(sq * (1.0f / 32.0f) - mu * mu + 1e-5f);
    const v4f gt0 = *(const v4f*)(g + mofs);
    const v4f gt1 = *(const v4f*)(g + mofs + 4);
    const v4f go0 = *(const v4f*)(g + 16 + mofs);
    const v4f go1 = *(const v4f*)(g + 16 + mofs + 4);
    const v4f at0 = *(const v4f*)(bb + mofs);
    const v4f at1 = *(const v4f*)(bb + mofs + 4);
    const v4f ao0 = *(const v4f*)(bb + 16 + mofs);
    const v4f ao1 = *(const v4f*)(bb + 16 + mofs + 4);
#pragma unroll
    for (int r = 0; r < 4; ++r) {
      vt[r] = (vt[r] - mu) * rs * gt0[r] + at0[r];
      vt[4 + r] = (vt[4 + r] - mu) * rs * gt1[r] + at1[r];
      vb[r] = (vb[r] - mu) * rs * go0[r] + ao0[r];
      vb[4 + r] = (vb[4 + r] - mu) * rs * go1[r] + ao1[r];
    }
  }
  // convert to f16 dwords (register-only) and exchange halves with lane^16:
  // lane<16 keeps top (feats 0..7), sends bottom (16..23), receives partner
  // top (8..15); lane>=16 mirrored.
  int T[4], U[4];
#pragma unroll
  for (int q = 0; q < 4; ++q) {
    T[q] = pkh(vt[2 * q], vt[2 * q + 1]);
    U[q] = pkh(vb[2 * q], vb[2 * q + 1]);
  }
  const bool lo = (lane < 16);
  v8i xi;
#pragma unroll
  for (int q = 0; q < 4; ++q) {
    const int send = lo ? U[q] : T[q];
    const int got = xl16(send);  // v_permlanex16_b32
    xi[q] = lo ? T[q] : got;
    xi[4 + q] = lo ? got : U[q];
  }
  return __builtin_bit_cast(v16h, xi);
}

// Full fused MLP on a wave's 16 elements, activations resident in registers.
// Returns output C tile: lane holds elem n0=lane&15, feats mofs..mofs+7.
template <bool LN>
__device__ __forceinline__ v8f wave_mlp(int ktiles, int dout, v16h x0, v16h x1,
                                        const _Float16* __restrict__ winf,
                                        const float* __restrict__ bin,
                                        const _Float16* __restrict__ whf,
                                        const float* __restrict__ bh,
                                        const _Float16* __restrict__ woutf,
                                        const float* __restrict__ bout,
                                        const float* __restrict__ lng,
                                        const float* __restrict__ lnb,
                                        int lane) {
  __builtin_prefetch(whf, 0, 3);  // global_prefetch toward WGP
  // ---- input layer (K tiles accumulate through the C operand)
  v8f ct = {}, cb = {};
  {
    const v16h a0 = load_frag(winf, 0, lane);
    const v16h a1 = load_frag(winf, 1, lane);
    ct = wmma16x16x32(a0, x0, ct);
    cb = wmma16x16x32(a1, x0, cb);
    if (ktiles == 2) {
      const v16h a2 = load_frag(winf, 2, lane);
      const v16h a3 = load_frag(winf, 3, lane);
      ct = wmma16x16x32(a2, x1, ct);
      cb = wmma16x16x32(a3, x1, cb);
    }
  }
  v16h x = epilogue<LN>(ct, cb, bin, lng, lnb, lane);
  // ---- 32 hidden layers: 4x global_load_b128 + 2x WMMA + register epilogue
  for (int l = 0; l < NL_; ++l) {
    const v16h a0 = load_frag(whf, l * 2, lane);
    const v16h a1 = load_frag(whf, l * 2 + 1, lane);
    const v8f z = {};
    const v8f d0 = wmma16x16x32(a0, x, z);
    const v8f d1 = wmma16x16x32(a1, x, z);
    x = epilogue<LN>(d0, d1, bh + (size_t)l * H_, lng + (size_t)(l + 1) * H_,
                     lnb + (size_t)(l + 1) * H_, lane);
  }
  // ---- output layer (dout <= 16): single A tile
  const v16h a = load_frag(woutf, 0, lane);
  const v8f z = {};
  v8f o = wmma16x16x32(a, x, z);
  const int mofs = (lane < 16) ? 0 : 8;
#pragma unroll
  for (int r = 0; r < 8; ++r) {
    const int f = mofs + r;
    const float ob = (f < dout) ? bout[f] : 0.0f;
    o[r] = fmaxf(o[r] + ob, 0.0f);
  }
  return o;
}

// Store 8 consecutive features of element n0: two global_store_b128.
__device__ __forceinline__ void write_out16(float* __restrict__ dst,
                                            size_t elem, v8f o, int lane) {
  const int mofs = (lane < 16) ? 0 : 8;
  float* p = dst + elem * 16 + mofs;
  v4f o0, o1;
#pragma unroll
  for (int r = 0; r < 4; ++r) {
    o0[r] = o[r];
    o1[r] = o[4 + r];
  }
  *(v4f*)p = o0;
  *(v4f*)(p + 4) = o1;
}

#define WAVE_IDS                                                               \
  const int lane = threadIdx.x & 31;                                           \
  const int wave = threadIdx.x >> 5;                                           \
  const int n0 = lane & 15;                                                    \
  const size_t base = (size_t)blockIdx.x * 128 + (size_t)wave * 16;            \
  const size_t ge = base + n0; /* this lane's element (2 lanes per element) */

// ---------------- node encoder: x=[nodes(6),g] -> ve(16) ------------------
__global__ __launch_bounds__(TPB) void k_encode_nodes(
    const float* __restrict__ nodes, const float* __restrict__ gvec,
    const _Float16* winf, const float* bin, const _Float16* whf,
    const float* bh, const _Float16* woutf, const float* bout,
    float* __restrict__ ve) {
  WAVE_IDS
  v16h x0 = {};  // lane<16: feats 0..15, lane>=16: feats 16..31 (all zero)
  if (lane < 16) {
    const int b = (int)(ge / N_);
    const float* np = nodes + ge * 6;
#pragma unroll
    for (int k = 0; k < 6; ++k) x0[k] = (_Float16)np[k];
    x0[6] = (_Float16)gvec[b];
  }
  const v16h x1 = {};
  const v8f o = wave_mlp<false>(1, 16, x0, x1, winf, bin, whf, bh, woutf, bout,
                                bin, bin, lane);
  write_out16(ve, ge, o, lane);
}

// ---------------- edge encoder: [edge(1), rel(3), |rel|(1)] -> ee(16) -----
__global__ __launch_bounds__(TPB) void k_encode_edges(
    const float* __restrict__ nodes, const float* __restrict__ edges,
    const int* __restrict__ senders, const int* __restrict__ receivers,
    const _Float16* winf, const float* bin, const _Float16* whf,
    const float* bh, const _Float16* woutf, const float* bout,
    float* __restrict__ ee) {
  WAVE_IDS
  v16h x0 = {};
  if (lane < 16) {
    const int b = (int)(ge / E_);
    const int s = senders[ge];
    const int rc = receivers[ge];
    const float* ps = nodes + ((size_t)b * N_ + s) * 6;
    const float* pr = nodes + ((size_t)b * N_ + rc) * 6;
    const float rx = ps[0] - pr[0], ry = ps[1] - pr[1], rz = ps[2] - pr[2];
    x0[0] = (_Float16)edges[ge];
    x0[1] = (_Float16)rx;
    x0[2] = (_Float16)ry;
    x0[3] = (_Float16)rz;
    x0[4] = (_Float16)sqrtf(rx * rx + ry * ry + rz * rz);
  }
  const v16h x1 = {};
  const v8f o = wave_mlp<false>(1, 16, x0, x1, winf, bin, whf, bh, woutf, bout,
                                bin, bin, lane);
  write_out16(ee, ge, o, lane);
}

// --------- edge update (pe, LN): [ee,vs,vr,g](49) -> ee, scatter-add agg --
__global__ __launch_bounds__(TPB) void k_edge_update(
    const float* __restrict__ gvec, const int* __restrict__ senders,
    const int* __restrict__ receivers, const float* __restrict__ ve,
    const _Float16* winf, const float* bin, const _Float16* whf,
    const float* bh, const _Float16* woutf, const float* bout,
    const float* lng, const float* lnb, float* __restrict__ ee,
    float* __restrict__ agg) {
  WAVE_IDS
  const int b = (int)(ge / E_);
  const int s = senders[ge];
  const int rc = receivers[ge];
  const size_t aggrow = ((size_t)b * N_ + rc) * 16;
  // frag0: feats 0..15 = ee row (lane<16) | feats 16..31 = ve[sender] (lane>=16)
  const float* p0 =
      (lane < 16) ? (ee + ge * 16) : (ve + ((size_t)b * N_ + s) * 16);
  const v16h x0 = load16f(p0);
  // frag1: feats 32..47 = ve[receiver] (lane<16) | feat 48 = g (lane>=16)
  v16h x1 = {};
  if (lane < 16) {
    x1 = load16f(ve + ((size_t)b * N_ + rc) * 16);
  } else {
    x1[0] = (_Float16)gvec[b];
  }
  const v8f o = wave_mlp<true>(2, 16, x0, x1, winf, bin, whf, bh, woutf, bout,
                               lng, lnb, lane);
  write_out16(ee, ge, o, lane);
  const int mofs = (lane < 16) ? 0 : 8;
#pragma unroll
  for (int r = 0; r < 8; ++r)
    atomicAdd(&agg[aggrow + mofs + r], o[r]);  // segment_sum over receivers
}

// --------- node update (pv, LN): [agg,ve,g](33) -> ve ---------------------
__global__ __launch_bounds__(TPB) void k_node_update(
    const float* __restrict__ gvec, const float* __restrict__ agg,
    const _Float16* winf, const float* bin, const _Float16* whf,
    const float* bh, const _Float16* woutf, const float* bout,
    const float* lng, const float* lnb, float* __restrict__ ve) {
  WAVE_IDS
  const int b = (int)(ge / N_);
  const float* p0 = (lane < 16) ? (agg + ge * 16) : (ve + ge * 16);
  const v16h x0 = load16f(p0);
  v16h x1 = {};
  if (lane < 16) x1[0] = (_Float16)gvec[b];  // feat 32
  const v8f o = wave_mlp<true>(2, 16, x0, x1, winf, bin, whf, bh, woutf, bout,
                               lng, lnb, lane);
  write_out16(ve, ge, o, lane);  // safe: depends only on own element
}

// ---------------- decoder: ve(16) -> out(3) -------------------------------
__global__ __launch_bounds__(TPB) void k_decode(
    const float* __restrict__ ve, const _Float16* winf, const float* bin,
    const _Float16* whf, const float* bh, const _Float16* woutf,
    const float* bout, float* __restrict__ out) {
  WAVE_IDS
  v16h x0 = {};
  if (lane < 16) x0 = load16f(ve + ge * 16);
  const v16h x1 = {};
  const v8f o = wave_mlp<false>(1, 3, x0, x1, winf, bin, whf, bh, woutf, bout,
                                bin, bin, lane);
  if (lane < 16) {  // feats 0..2 live in the mofs=0 half
#pragma unroll
    for (int r = 0; r < 3; ++r) out[ge * 3 + r] = o[r];
  }
}

static inline void pack_w(const float* W, _Float16* dst, int ld, int ktotal,
                          int nmax, int nfrag, hipStream_t s) {
  const int threads = nfrag * 32;
  const int blocks = (threads + TPB - 1) / TPB;
  k_pack<<<blocks, TPB, 0, s>>>(W, dst, ld, ktotal, nmax, nfrag);
}

extern "C" void kernel_launch(void* const* d_in, const int* in_sizes, int n_in,
                              void* d_out, int out_size, void* d_ws,
                              size_t ws_size, hipStream_t stream) {
  (void)in_sizes; (void)n_in; (void)out_size; (void)ws_size;
#define F(i) ((const float*)d_in[(i)])
#define I(i) ((const int*)d_in[(i)])
  const float* nodes = F(0);
  const float* edges = F(1);
  const float* gvec = F(2);
  const int* senders = I(3);
  const int* receivers = I(4);
  // ne: 5..10 | ed: 11..16 | pe: 17..24 (+lng/lnb) | pv: 25..32 | de: 33..38
  float* ve = (float*)d_ws;                    // B*N*16 f32 (4 MB)
  float* ee = ve + (size_t)B_ * N_ * 16;       // B*E*16 f32 (4 MB)
  float* agg = ee + (size_t)B_ * E_ * 16;      // B*N*16 f32 (4 MB)
  _Float16* wp = (_Float16*)(agg + (size_t)B_ * N_ * 16);  // packed W^T frags

  _Float16* ne_p = wp + 0 * (size_t)MS;
  _Float16* ed_p = wp + 1 * (size_t)MS;
  _Float16* pe_p = wp + 2 * (size_t)MS;
  _Float16* pv_p = wp + 3 * (size_t)MS;
  _Float16* de_p = wp + 4 * (size_t)MS;

  // ---- pack all weights to f16 A-fragment (W^T) order --------------------
  pack_w(F(5), ne_p, 32, 7, 32, 2, stream);                       // ne_Win
  pack_w(F(7), ne_p + 2048, 32, 32 * NL_, 32, 2 * NL_, stream);   // ne_Wh
  pack_w(F(9), ne_p + 34816, 16, 32, 16, 1, stream);              // ne_Wout
  pack_w(F(11), ed_p, 32, 5, 32, 2, stream);                      // ed_Win
  pack_w(F(13), ed_p + 2048, 32, 32 * NL_, 32, 2 * NL_, stream);  // ed_Wh
  pack_w(F(15), ed_p + 34816, 16, 32, 16, 1, stream);             // ed_Wout
  pack_w(F(17), pe_p, 32, 49, 32, 4, stream);                     // pe_Win
  pack_w(F(19), pe_p + 2048, 32, 32 * NL_, 32, 2 * NL_, stream);  // pe_Wh
  pack_w(F(21), pe_p + 34816, 16, 32, 16, 1, stream);             // pe_Wout
  pack_w(F(25), pv_p, 32, 33, 32, 4, stream);                     // pv_Win
  pack_w(F(27), pv_p + 2048, 32, 32 * NL_, 32, 2 * NL_, stream);  // pv_Wh
  pack_w(F(29), pv_p + 34816, 16, 32, 16, 1, stream);             // pv_Wout
  pack_w(F(33), de_p, 32, 16, 32, 2, stream);                     // de_Win
  pack_w(F(35), de_p + 2048, 32, 32 * NL_, 32, 2 * NL_, stream);  // de_Wh
  pack_w(F(37), de_p + 34816, 3, 32, 3, 1, stream);               // de_Wout

  const dim3 blk(TPB);
  const int gn = (B_ * N_) / 128;  // 512
  const int gE = (B_ * E_) / 128;  // 512

  k_encode_nodes<<<gn, blk, 0, stream>>>(nodes, gvec, ne_p, F(6), ne_p + 2048,
                                         F(8), ne_p + 34816, F(10), ve);
  k_encode_edges<<<gE, blk, 0, stream>>>(nodes, edges, senders, receivers,
                                         ed_p, F(12), ed_p + 2048, F(14),
                                         ed_p + 34816, F(16), ee);
  for (int step = 0; step < 2; ++step) {
    hipMemsetAsync(agg, 0, (size_t)B_ * N_ * 16 * sizeof(float), stream);
    k_edge_update<<<gE, blk, 0, stream>>>(gvec, senders, receivers, ve, pe_p,
                                          F(18), pe_p + 2048, F(20),
                                          pe_p + 34816, F(22), F(23), F(24),
                                          ee, agg);
    k_node_update<<<gn, blk, 0, stream>>>(gvec, agg, pv_p, F(26), pv_p + 2048,
                                          F(28), pv_p + 34816, F(30), F(31),
                                          F(32), ve);
  }
  k_decode<<<gn, blk, 0, stream>>>(ve, de_p, F(34), de_p + 2048, F(36),
                                   de_p + 34816, F(38), (float*)d_out);
#undef F
#undef I
}
